// Road_Emb_87720412054015
// MI455X (gfx1250) — compile-verified, matrix operand order checked by reading, not run on hardware
//
#include <hip/hip_runtime.h>
#include <math.h>

// ---------------------------------------------------------------------------
// Types for CDNA5 WMMA (wave32, gfx1250)
// ---------------------------------------------------------------------------
typedef __bf16 bf16;
typedef bf16  v16bf __attribute__((ext_vector_type(16)));
typedef bf16  bf16x4 __attribute__((ext_vector_type(4)));
typedef float v8f   __attribute__((ext_vector_type(8)));

#define GF_BIAS    1
#define GF_ACCUM   2
#define GF_OUTBF16 4

// K-offset within a 32-wide K-step for fragment element pair j (j=0..7),
// per the ISA 16-bit A/B fragment layout:
//   lanes 0-15 : K = {0..7, 16..23};  lanes 16-31 : K = {8..15, 24..31}
__device__ __forceinline__ int kb_of(int j) {
    return (j < 4) ? 2 * j : 16 + 2 * (j - 4);
}

// ---------------------------------------------------------------------------
// Pack row-major f32 B[K,Nout] into per-lane WMMA B-fragment bf16 layout:
//   P[((t*(K/32)+s)*32 + lane)*16 + e]
// ---------------------------------------------------------------------------
__global__ void __launch_bounds__(256)
pack_b_kernel(const float* __restrict__ B, bf16* __restrict__ P, int K, int Nout)
{
    int id = blockIdx.x * blockDim.x + threadIdx.x;
    int ksteps = K >> 5;
    int total = (Nout >> 4) * ksteps * 32;
    if (id >= total) return;
    int lane = id & 31;
    int rest = id >> 5;
    int s = rest % ksteps;          // k-step
    int t = rest / ksteps;          // 16-wide N tile
    int half = lane >> 4, ls = lane & 15;
    int col = t * 16 + ls;
    int k0 = s * 32 + (half ? 8 : 0);
    bf16* o = P + id * 16;
#pragma unroll
    for (int j = 0; j < 8; ++j) {
        int k = k0 + kb_of(j);
        o[2 * j]     = (bf16)B[k * Nout + col];
        o[2 * j + 1] = (bf16)B[(k + 1) * Nout + col];
    }
}

// ---------------------------------------------------------------------------
// Pack row-major f32 A[M,K] into per-lane WMMA A-fragment bf16 layout:
//   P[((tm*(K/32)+s)*32 + lane)*16 + e]
// Lane's 16 values are two contiguous 8-float runs -> 4x float4 + cvt.
// M % 16 == 0, K % 32 == 0.
// ---------------------------------------------------------------------------
__global__ void __launch_bounds__(256)
pack_a_kernel(const float* __restrict__ A, bf16* __restrict__ P, int M, int K)
{
    int id = blockIdx.x * blockDim.x + threadIdx.x;
    int ksteps = K >> 5;
    int total = (M >> 4) * ksteps * 32;
    if (id >= total) return;
    int lane = id & 31;
    int rest = id >> 5;
    int s = rest % ksteps;
    int tm = rest / ksteps;
    int half = lane >> 4, ls = lane & 15;
    int row = tm * 16 + ls;
    int aoff = row * K + s * 32 + (half ? 8 : 0);
    const float4* ap = reinterpret_cast<const float4*>(A + aoff);
    float4 x0 = ap[0];
    float4 x1 = ap[1];
    float4 x2 = ap[4];                 // +16 floats
    float4 x3 = ap[5];
    v16bf o;
    o[0]=(bf16)x0.x; o[1]=(bf16)x0.y; o[2]=(bf16)x0.z; o[3]=(bf16)x0.w;
    o[4]=(bf16)x1.x; o[5]=(bf16)x1.y; o[6]=(bf16)x1.z; o[7]=(bf16)x1.w;
    o[8]=(bf16)x2.x; o[9]=(bf16)x2.y; o[10]=(bf16)x2.z; o[11]=(bf16)x2.w;
    o[12]=(bf16)x3.x; o[13]=(bf16)x3.y; o[14]=(bf16)x3.z; o[15]=(bf16)x3.w;
    *reinterpret_cast<v16bf*>(P + id * 16) = o;
}

// ---------------------------------------------------------------------------
// WMMA GEMM: C[M,Nout] = A x B.  B pre-packed bf16 fragments; A either
// pre-packed bf16 fragments (PACKEDA=true) or raw f32 row-major (converted
// in-register).  Compile-time specialization keeps the K-loop branch-free:
// the packed-A variant is pure vmem(b128) + wmma.
// K % 32 == 0, Nout % 64 == 0, M % 16 == 0.
// One wave32 computes a 16x64 strip: 4 WMMAs / K-step, A fragment reused 4x.
// ---------------------------------------------------------------------------
template <bool PACKEDA>
__global__ void __launch_bounds__(128)
wmma_gemm_bf16(const float* __restrict__ A, const bf16* __restrict__ PA,
               const bf16* __restrict__ P, const float* __restrict__ bias,
               float* __restrict__ C, bf16* __restrict__ Cb,
               int M, int K, int Nout, int ldc, int col0,
               int stripsN, int totalStrips, int flags)
{
    const int gtid = blockIdx.x * blockDim.x + threadIdx.x;
    const int wave = gtid >> 5;
    if (wave >= totalStrips) return;           // wave-uniform (EXEC stays full)
    const int lane = gtid & 31;
    const int sM = wave / stripsN;
    const int sN = wave % stripsN;             // 64-wide strip
    const int half = lane >> 4;
    const int ls   = lane & 15;

    const int ksteps = K >> 5;
    const int tile0 = sN * 4;                  // first 16-wide N tile of strip

    v8f acc0 = {0.f,0.f,0.f,0.f,0.f,0.f,0.f,0.f};
    v8f acc1 = acc0, acc2 = acc0, acc3 = acc0;

    // per-lane base offsets (all fit in 32-bit)
    int aoff  = (sM * 16 + ls) * K + (half ? 8 : 0);       // raw-A path
    int paoff = (sM * ksteps * 32 + lane) * 16;            // packed-A path
    int pb0 = ((tile0 + 0) * ksteps * 32 + lane) * 16;
    int pb1 = ((tile0 + 1) * ksteps * 32 + lane) * 16;
    int pb2 = ((tile0 + 2) * ksteps * 32 + lane) * 16;
    int pb3 = ((tile0 + 3) * ksteps * 32 + lane) * 16;

#pragma unroll 2
    for (int s = 0; s < ksteps; ++s) {
        v16bf af;
        if constexpr (PACKEDA) {
            af = *reinterpret_cast<const v16bf*>(PA + paoff);    // 2x b128
        } else {
            const float4* ap = reinterpret_cast<const float4*>(A + aoff);
            float4 x0 = ap[0];
            float4 x1 = ap[1];
            float4 x2 = ap[4];                 // +16 floats
            float4 x3 = ap[5];
            af[0]=(bf16)x0.x; af[1]=(bf16)x0.y; af[2]=(bf16)x0.z; af[3]=(bf16)x0.w;
            af[4]=(bf16)x1.x; af[5]=(bf16)x1.y; af[6]=(bf16)x1.z; af[7]=(bf16)x1.w;
            af[8]=(bf16)x2.x; af[9]=(bf16)x2.y; af[10]=(bf16)x2.z; af[11]=(bf16)x2.w;
            af[12]=(bf16)x3.x; af[13]=(bf16)x3.y; af[14]=(bf16)x3.z; af[15]=(bf16)x3.w;
        }

        // B fragments: one contiguous 32B v16bf load each
        v16bf b0 = *reinterpret_cast<const v16bf*>(P + pb0);
        v16bf b1 = *reinterpret_cast<const v16bf*>(P + pb1);
        v16bf b2 = *reinterpret_cast<const v16bf*>(P + pb2);
        v16bf b3 = *reinterpret_cast<const v16bf*>(P + pb3);

        acc0 = __builtin_amdgcn_wmma_f32_16x16x32_bf16(false, af, false, b0, (short)0, acc0, false, false);
        acc1 = __builtin_amdgcn_wmma_f32_16x16x32_bf16(false, af, false, b1, (short)0, acc1, false, false);
        acc2 = __builtin_amdgcn_wmma_f32_16x16x32_bf16(false, af, false, b2, (short)0, acc2, false, false);
        acc3 = __builtin_amdgcn_wmma_f32_16x16x32_bf16(false, af, false, b3, (short)0, acc3, false, false);

        aoff += 32;
        paoff += 512;
        pb0 += 512; pb1 += 512; pb2 += 512; pb3 += 512;   // 32 lanes * 16 elems
    }

    // Epilogue. C/D layout: VGPR r -> row sM*16 + r + (half?8:0), col per lane.
    const int orow0 = sM * 16 + (half ? 8 : 0);
#pragma unroll
    for (int t = 0; t < 4; ++t) {
        v8f a = (t == 0) ? acc0 : (t == 1) ? acc1 : (t == 2) ? acc2 : acc3;
        int ocol = (tile0 + t) * 16 + ls;
        float bv = (flags & GF_BIAS) ? bias[ocol] : 0.f;
#pragma unroll
        for (int r = 0; r < 8; ++r) {
            int oidx = (orow0 + r) * ldc + col0 + ocol;
            float v = a[r] + bv;
            if (flags & GF_ACCUM) v += C[oidx];
            if (flags & GF_OUTBF16) Cb[oidx] = (bf16)v;
            else                    C[oidx]  = v;
        }
    }
}

// ---------------------------------------------------------------------------
// poi projection (K=12): negligible FLOPs, plain VALU kernel.
// ---------------------------------------------------------------------------
__global__ void __launch_bounds__(256)
poi_proj_kernel(const float* __restrict__ X, const float* __restrict__ W,
                const float* __restrict__ bias, float* __restrict__ C, int N)
{
    int id = blockIdx.x * blockDim.x + threadIdx.x;
    if (id >= N * 64) return;
    int n = id >> 6, c = id & 63;
    float acc = bias[c];
#pragma unroll
    for (int k = 0; k < 12; ++k) acc += X[n * 12 + k] * W[k * 64 + c];
    C[n * 256 + c] = acc;
}

// ---------------------------------------------------------------------------
// Edge gather/scale/scatter: one wave32 per edge.
//   agg[dst] += sum_b coeff[etype,b] * tb[b][src]     (D = 256)
// tb bf16 [8,N,256] (~82 MB -> resident in 192 MB L2).
// ---------------------------------------------------------------------------
__global__ void __launch_bounds__(256)
edge_scatter(const bf16* __restrict__ tb,
             const int* __restrict__ src, const int* __restrict__ dst,
             const int* __restrict__ ety, const float* __restrict__ coeff,
             float* __restrict__ agg, int E, int N)
{
    const int gtid = blockIdx.x * blockDim.x + threadIdx.x;
    const int e = gtid >> 5;
    const int lane = gtid & 31;
    if (e >= E) return;

    const int s = src[e];
    const int d = dst[e];
    const int t = ety[e];

    float w[8];
#pragma unroll
    for (int b = 0; b < 8; ++b) w[b] = coeff[t * 8 + b];

    const int nd = N * 256;                 // < 2^31
    const int srow = s * 256;
    const int drow = d * 256;

    __builtin_prefetch(tb + srow + 128 + lane * 4, 0, 1);

#pragma unroll
    for (int it = 0; it < 2; ++it) {
        const int col = it * 128 + lane * 4;
        float a0 = 0.f, a1 = 0.f, a2 = 0.f, a3 = 0.f;
#pragma unroll
        for (int b = 0; b < 8; ++b) {
            bf16x4 v = *reinterpret_cast<const bf16x4*>(tb + b * nd + srow + col);
            a0 += w[b] * (float)v[0];
            a1 += w[b] * (float)v[1];
            a2 += w[b] * (float)v[2];
            a3 += w[b] * (float)v[3];
        }
        float* o = agg + drow + col;
        atomicAdd(o + 0, a0);
        atomicAdd(o + 1, a1);
        atomicAdd(o + 2, a2);
        atomicAdd(o + 3, a3);
    }
}

// ---------------------------------------------------------------------------
// Elementwise kernels
// ---------------------------------------------------------------------------
__global__ void relu_kernel(float* __restrict__ x, int n)
{
    int i = blockIdx.x * blockDim.x + threadIdx.x;
    if (i < n) x[i] = fmaxf(x[i], 0.f);
}

__global__ void gelu_kernel(float* __restrict__ x, int n)
{
    int i = blockIdx.x * blockDim.x + threadIdx.x;
    if (i < n) {
        float v = x[i];
        x[i] = 0.5f * v * (1.0f + erff(v * 0.70710678118654752440f));
    }
}

// ---------------------------------------------------------------------------
// Host-side orchestration
// ---------------------------------------------------------------------------
extern "C" void kernel_launch(void* const* d_in, const int* in_sizes, int n_in,
                              void* d_out, int out_size, void* d_ws, size_t ws_size,
                              hipStream_t stream)
{
    (void)n_in; (void)ws_size; (void)out_size;
    const float* poi    = (const float*)d_in[0];
    const float* timeD  = (const float*)d_in[1];
    const float* road   = (const float*)d_in[2];
    const float* W_poi  = (const float*)d_in[3];
    const float* b_poi  = (const float*)d_in[4];
    const float* W_time = (const float*)d_in[5];
    const float* b_time = (const float*)d_in[6];
    const float* W_road = (const float*)d_in[7];
    const float* b_road = (const float*)d_in[8];

    const float* coeff1[2]  = {(const float*)d_in[9],  (const float*)d_in[17]};
    const float* bases1[2]  = {(const float*)d_in[10], (const float*)d_in[18]};
    const float* loop1[2]   = {(const float*)d_in[11], (const float*)d_in[19]};
    const float* bias1[2]   = {(const float*)d_in[12], (const float*)d_in[20]};
    const float* coeff2[2]  = {(const float*)d_in[13], (const float*)d_in[21]};
    const float* bases2[2]  = {(const float*)d_in[14], (const float*)d_in[22]};
    const float* loop2[2]   = {(const float*)d_in[15], (const float*)d_in[23]};
    const float* bias2[2]   = {(const float*)d_in[16], (const float*)d_in[24]};

    const float* W_fuse    = (const float*)d_in[25];
    const float* b_fuse    = (const float*)d_in[26];
    const float* W_fuse_kg = (const float*)d_in[27];
    const float* b_fuse_kg = (const float*)d_in[28];

    const int* esrc[2] = {(const int*)d_in[29], (const int*)d_in[32]};
    const int* edst[2] = {(const int*)d_in[30], (const int*)d_in[33]};
    const int* etyp[2] = {(const int*)d_in[31], (const int*)d_in[34]};

    const int N = in_sizes[0] / 12;       // 20000
    const int E = in_sizes[29];           // 320000
    const int K_time = in_sizes[1] / N;   // 2880
    const int D = 256;
    const int B = 8;
    const size_t ND = (size_t)N * D;

    float* out = (float*)d_out;

    // ---- workspace carve-out -------------------------------------------
    char* ws = (char*)d_ws;
    float* feat = (float*)ws;  ws += ND * sizeof(float);            // 20.5 MB
    bf16*  tb   = (bf16*)ws;   ws += (size_t)B * ND * sizeof(bf16); // 82 MB
    float* t0   = (float*)ws;  ws += ND * sizeof(float);
    float* t1   = (float*)ws;  ws += ND * sizeof(float);
    float* t2   = (float*)ws;  ws += ND * sizeof(float);
    bf16*  paF  = (bf16*)ws;   ws += ND * sizeof(bf16);  // packed A (feat / t1)
    bf16*  paT  = (bf16*)ws;   ws += ND * sizeof(bf16);  // packed A (t0 / t2)
    // packed bf16 weights
    bf16* p_time   = (bf16*)ws; ws += (size_t)K_time * 128 * sizeof(bf16);
    bf16* p_road   = (bf16*)ws; ws += (size_t)32 * 64 * sizeof(bf16);
    bf16* p_b1[2], *p_l1[2], *p_b2[2], *p_l2[2];
    for (int g = 0; g < 2; ++g) {
        p_b1[g] = (bf16*)ws; ws += (size_t)B * D * D * sizeof(bf16);
        p_l1[g] = (bf16*)ws; ws += (size_t)D * D * sizeof(bf16);
        p_b2[g] = (bf16*)ws; ws += (size_t)B * D * D * sizeof(bf16);
        p_l2[g] = (bf16*)ws; ws += (size_t)D * D * sizeof(bf16);
    }
    bf16* p_fuse   = (bf16*)ws; ws += (size_t)D * D * sizeof(bf16);
    bf16* p_fusekg = (bf16*)ws; ws += (size_t)D * D * sizeof(bf16);

    auto packB = [&](const float* Bm, bf16* P, int K, int Nout) {
        int total = (Nout >> 4) * (K >> 5) * 32;
        pack_b_kernel<<<(total + 255) / 256, 256, 0, stream>>>(Bm, P, K, Nout);
    };
    auto packA = [&](const float* Am, bf16* P, int M, int K) {
        int total = (M >> 4) * (K >> 5) * 32;
        pack_a_kernel<<<(total + 255) / 256, 256, 0, stream>>>(Am, P, M, K);
    };

    auto gemm_raw = [&](const float* A, const bf16* P, const float* bias,
                        float* C, int M, int K, int Nout,
                        int ldc, int col0, int flags) {
        int strips = (M / 16) * (Nout / 64);
        int blocks = (strips + 3) / 4;          // 4 waves / block (128 thr)
        wmma_gemm_bf16<false><<<blocks, 128, 0, stream>>>(
            A, nullptr, P, bias, C, nullptr, M, K, Nout, ldc, col0,
            Nout / 64, strips, flags);
    };
    auto gemm_pk = [&](const bf16* PA, const bf16* P, const float* bias,
                       float* C, bf16* Cb, int M, int K, int Nout,
                       int ldc, int col0, int flags) {
        int strips = (M / 16) * (Nout / 64);
        int blocks = (strips + 3) / 4;
        wmma_gemm_bf16<true><<<blocks, 128, 0, stream>>>(
            nullptr, PA, P, bias, C, Cb, M, K, Nout, ldc, col0,
            Nout / 64, strips, flags);
    };

    // ---- pack all weight matrices once ---------------------------------
    packB(W_time, p_time, K_time, 128);
    packB(W_road, p_road, 32, 64);
    for (int g = 0; g < 2; ++g) {
        for (int b = 0; b < B; ++b) {
            packB(bases1[g] + (size_t)b * D * D, p_b1[g] + (size_t)b * D * D, D, D);
            packB(bases2[g] + (size_t)b * D * D, p_b2[g] + (size_t)b * D * D, D, D);
        }
        packB(loop1[g], p_l1[g], D, D);
        packB(loop2[g], p_l2[g], D, D);
    }
    packB(W_fuse,    p_fuse,   D, D);
    packB(W_fuse_kg, p_fusekg, D, D);

    // ---- input projections -> feat [N,256] = [poi64 | time128 | road64] --
    // time/road: single-use A, raw f32 path (packing would double traffic)
    poi_proj_kernel<<<(N * 64 + 255) / 256, 256, 0, stream>>>(poi, W_poi, b_poi, feat, N);
    gemm_raw(timeD, p_time, b_time, feat, N, K_time, 128, D, 64,  GF_BIAS);
    gemm_raw(road,  p_road, b_road, feat, N, 32,     64,  D, 192, GF_BIAS);

    const int ew_blocks = (int)(((size_t)E * 32 + 255) / 256);
    const int el_blocks = (int)((ND + 255) / 256);

    // relconv with pre-packed A shared by all 9 GEMMs
    auto relconv = [&](const bf16* pa, const float* coeff, const bf16* pbases,
                       const bf16* ploop, const float* bias, int g, float* o) {
        for (int b = 0; b < B; ++b)
            gemm_pk(pa, pbases + (size_t)b * D * D, nullptr, nullptr,
                    tb + (size_t)b * ND, N, D, D, D, 0, GF_OUTBF16);
        gemm_pk(pa, ploop, bias, o, nullptr, N, D, D, D, 0, GF_BIAS);
        edge_scatter<<<ew_blocks, 256, 0, stream>>>(tb, esrc[g], edst[g], etyp[g],
                                                    coeff, o, E, N);
    };

    packA(feat, paF, N, D);          // shared by layer-1 of both branches

    // ---- adj branch (g=0) ----------------------------------------------
    relconv(paF, coeff1[0], p_b1[0], p_l1[0], bias1[0], 0, t0);
    relu_kernel<<<el_blocks, 256, 0, stream>>>(t0, (int)ND);
    packA(t0, paT, N, D);
    relconv(paT, coeff2[0], p_b2[0], p_l2[0], bias2[0], 0, t1);

    // ---- no_adj branch (g=1) -------------------------------------------
    relconv(paF, coeff1[1], p_b1[1], p_l1[1], bias1[1], 1, t0);
    relu_kernel<<<el_blocks, 256, 0, stream>>>(t0, (int)ND);
    packA(t0, paT, N, D);
    relconv(paT, coeff2[1], p_b2[1], p_l2[1], bias2[1], 1, t2);

    // ---- fusion + exact GELU -------------------------------------------
    packA(t1, paF, N, D);            // feat pack no longer needed
    gemm_pk(paF, p_fuse,   b_fuse,    out, nullptr, N, D, D, D, 0, GF_BIAS);
    packA(t2, paT, N, D);
    gemm_pk(paT, p_fusekg, b_fuse_kg, out, nullptr, N, D, D, D, 0,
            GF_BIAS | GF_ACCUM);
    gelu_kernel<<<el_blocks, 256, 0, stream>>>(out, (int)ND);
}
